// MultiHeadDotProductAttention_76605036691621
// MI455X (gfx1250) — matrix operand order repaired and checked by reading, not compiled, not run on hardware
//
#include <hip/hip_runtime.h>
#include <hip/hip_bf16.h>

// MHA forward for MI455X (gfx1250, wave32).
// All four GEMMs + both attention contractions run on v_wmma_f32_16x16x32_f16
// (f32 accumulate). Flash-attention online softmax: the [B,H,N,N] score matrix
// never touches memory. Softmax runs in the exp2 domain: SCALE*log2(e) is
// folded into the Q projection epilogue so the attention inner loop is pure
// v_exp_f32 with no extra multiplies.
//
// sched_group_barrier hints (DS-read group -> WMMA group) keep the fragment
// loads batched so the compiler emits partial s_wait_dscnt instead of a full
// LDS round-trip before every WMMA.
//
// Workspace layout (f16 elements), PH = B*H*N*c = 8388608:
//   Q  [B,H,N,64]   : ws[0    .. PH)     (pre-scaled by SCALE*log2e)
//   K  [B,H,N,64]   : ws[PH   .. 2PH)
//   V  [B,H,N,64]   : ws[2PH  .. 3PH)
//   O  [B,N,C]      : ws[3PH  .. 4PH)    (total 64 MB)

typedef __attribute__((ext_vector_type(16))) _Float16 v16h;
typedef __attribute__((ext_vector_type(8)))  _Float16 v8h;
typedef __attribute__((ext_vector_type(8)))  float    v8f;

#define DIMC       1024
#define SEQ_N      2048
#define NBATCH     4
#define NHEADS     16
#define HEADC      64
#define QSCALE_LOG2E (0.125f * 1.44269504088896f)   // 1/sqrt(64) * log2(e)

// sched_group_barrier masks
#define SGB_WMMA    0x008
#define SGB_DSREAD  0x100

static __device__ __forceinline__ v8f wmma_f16(v16h a, v16h b, v8f c) {
    return __builtin_amdgcn_wmma_f32_16x16x32_f16(
        /*neg_a=*/false, a, /*neg_b=*/false, b,
        /*c_mod=*/(short)0, c, /*reuse_a=*/false, /*reuse_b=*/false);
}

// A fragment (16x32, f16 source). ISA layout: lanes 0-15 = row (lane&15),
// elements 0..7 = K k0+0..7, 8..15 = K k0+16..23; lanes 16-31 shift K by +8.
static __device__ __forceinline__ v16h frag_a_f16(const _Float16* base, int ld,
                                                  int lane, int k0) {
    const _Float16* p = base + (size_t)(lane & 15) * ld + k0 + ((lane & 16) ? 8 : 0);
    v8h lo = *(const v8h*)p;
    v8h hi = *(const v8h*)(p + 16);
    v16h a;
#pragma unroll
    for (int i = 0; i < 8; ++i) { a[i] = lo[i]; a[8 + i] = hi[i]; }
    return a;
}

// A fragment built from f32 source with inline f32->f16 conversion.
static __device__ __forceinline__ v16h frag_a_f32(const float* base, int ld,
                                                  int lane, int k0) {
    const float* p = base + (size_t)(lane & 15) * ld + k0 + ((lane & 16) ? 8 : 0);
    const float4* q = (const float4*)p;
    float4 l0 = q[0], l1 = q[1], h0 = q[4], h1 = q[5];
    v16h a;
    a[0]  = (_Float16)l0.x; a[1]  = (_Float16)l0.y; a[2]  = (_Float16)l0.z; a[3]  = (_Float16)l0.w;
    a[4]  = (_Float16)l1.x; a[5]  = (_Float16)l1.y; a[6]  = (_Float16)l1.z; a[7]  = (_Float16)l1.w;
    a[8]  = (_Float16)h0.x; a[9]  = (_Float16)h0.y; a[10] = (_Float16)h0.z; a[11] = (_Float16)h0.w;
    a[12] = (_Float16)h1.x; a[13] = (_Float16)h1.y; a[14] = (_Float16)h1.z; a[15] = (_Float16)h1.w;
    return a;
}

// B fragment (32x16). Rows of Bs = output column n, contiguous along K.
// lanes 0-15: n = lane, K = 0..7,16..23; lanes 16-31: n = lane-16, K = 8..15,24..31.
static __device__ __forceinline__ v16h frag_b_lds(const _Float16* Bs, int ld, int lane) {
    const _Float16* p = Bs + (size_t)(lane & 15) * ld + ((lane & 16) ? 8 : 0);
    v8h lo = *(const v8h*)p;
    v8h hi = *(const v8h*)(p + 16);
    v16h b;
#pragma unroll
    for (int i = 0; i < 8; ++i) { b[i] = lo[i]; b[8 + i] = hi[i]; }
    return b;
}

static __device__ __forceinline__ void cvt_store_v8h(_Float16* dst, float4 u0, float4 u1) {
    v8h h;
    h[0] = (_Float16)u0.x; h[1] = (_Float16)u0.y; h[2] = (_Float16)u0.z; h[3] = (_Float16)u0.w;
    h[4] = (_Float16)u1.x; h[5] = (_Float16)u1.y; h[6] = (_Float16)u1.z; h[7] = (_Float16)u1.w;
    *(v8h*)dst = h;
}

// ---------------------------------------------------------------------------
// Kernel 1: QKV projection.  y = (x @ W.T + b) * outscale -> f16 [B,H,N,64].
// Block tile 128(M) x 64(N), K-steps of 32, double-buffered LDS W tile.
// blockIdx.z selects q/k/v; Q output carries SCALE*log2e.
// ---------------------------------------------------------------------------
__global__ __launch_bounds__(256)
void qkv_gemm_kernel(const float* __restrict__ x,
                     const float* __restrict__ wq, const float* __restrict__ bq,
                     const float* __restrict__ wk, const float* __restrict__ bk,
                     const float* __restrict__ wv, const float* __restrict__ bv,
                     _Float16* __restrict__ Q, _Float16* __restrict__ K,
                     _Float16* __restrict__ V) {
    const float* W; const float* bias; _Float16* Out; float outscale;
    if (blockIdx.z == 0)      { W = wq; bias = bq; Out = Q; outscale = QSCALE_LOG2E; }
    else if (blockIdx.z == 1) { W = wk; bias = bk; Out = K; outscale = 1.0f; }
    else                      { W = wv; bias = bv; Out = V; outscale = 1.0f; }

    __shared__ _Float16 Bs[2][64 * 40];           // double-buffered, pad 8
    const int tid = threadIdx.x, lane = tid & 31, w = tid >> 5;
    const int m0 = blockIdx.x * 128, n0 = blockIdx.y * 64;
    const float* Abase = x + (size_t)(m0 + w * 16) * DIMC;

    // This thread's staging slice of the W tile: row n, 8 K-contiguous floats.
    const int n = tid >> 2, koff = (tid & 3) << 3;
    const float* wsrc = W + (size_t)(n0 + n) * DIMC + koff;
    _Float16* const bdst = &Bs[0][0] + n * 40 + koff;

    {   // prologue: stage k-step 0 into buffer 0
        const float4* s4 = (const float4*)wsrc;
        cvt_store_v8h(bdst, s4[0], s4[1]);
    }

    v8f acc[4] = {};
    int cur = 0;
    for (int k0 = 0; k0 < DIMC; k0 += 32) {
        __syncthreads();                          // buffer `cur` ready
        const bool more = (k0 + 32) < DIMC;
        float4 u0, u1;
        if (more) {                               // global prefetch of next tile
            const float4* s4 = (const float4*)(wsrc + k0 + 32);
            u0 = s4[0]; u1 = s4[1];
        }
        // Hoist all fragment loads, then issue WMMAs back-to-back.
        v16h a = frag_a_f32(Abase, DIMC, lane, k0);
        v16h bf[4];
#pragma unroll
        for (int nt = 0; nt < 4; ++nt)
            bf[nt] = frag_b_lds(&Bs[cur][nt * 16 * 40], 40, lane);
#pragma unroll
        for (int nt = 0; nt < 4; ++nt)
            acc[nt] = wmma_f16(a, bf[nt], acc[nt]);
        // Pipeline hint: batch the 8 LDS reads, then the 4 WMMAs.
        __builtin_amdgcn_sched_group_barrier(SGB_DSREAD, 8, 0);
        __builtin_amdgcn_sched_group_barrier(SGB_WMMA,   4, 0);
        if (more) cvt_store_v8h(bdst + (cur ^ 1) * 64 * 40, u0, u1);
        cur ^= 1;
    }

    // Epilogue: (+ bias) * outscale, f16, scatter into [B, H, N, 64]
#pragma unroll
    for (int nt = 0; nt < 4; ++nt) {
        const int o  = n0 + nt * 16 + (lane & 15);
        const int hh = o >> 6, cc = o & 63;
        const float bval = bias[o];
#pragma unroll
        for (int i = 0; i < 8; ++i) {
            const int mrow = m0 + w * 16 + i + ((lane & 16) ? 8 : 0);
            const int bidx = mrow >> 11, qrow = mrow & (SEQ_N - 1);
            Out[(((size_t)bidx * NHEADS + hh) * SEQ_N + qrow) * HEADC + cc] =
                (_Float16)((acc[nt][i] + bval) * outscale);
        }
    }
}

// ---------------------------------------------------------------------------
// Kernel 2: flash attention. Block = 128 query rows of one (b,h); 8 waves,
// 16 query rows each. Key blocks of 64, online softmax in the exp2 domain
// (Q is pre-scaled by SCALE*log2e, so logits are already log2-domain).
// ---------------------------------------------------------------------------
__global__ __launch_bounds__(256)
void attn_kernel(const _Float16* __restrict__ Q, const _Float16* __restrict__ K,
                 const _Float16* __restrict__ V, _Float16* __restrict__ O) {
    __shared__ _Float16 Ks[64 * 72];        // [key][c]   stride 72
    __shared__ _Float16 Vt[64 * 72];        // [c][key]   stride 72 (transposed)
    __shared__ _Float16 Ps[8 * 16 * 72];    // per-wave P tile, [row][key]

    const int tid = threadIdx.x, lane = tid & 31, w = tid >> 5;
    const int bh = blockIdx.y;              // b*H + h
    const int b = bh >> 4, h = bh & 15;
    const int qb = blockIdx.x * 128 + w * 16;

    const _Float16* Qbase = Q + ((size_t)bh * SEQ_N + qb) * HEADC;
    const _Float16* Kbase = K + (size_t)bh * SEQ_N * HEADC;
    const _Float16* Vbase = V + (size_t)bh * SEQ_N * HEADC;
    _Float16* Pw = &Ps[w * 16 * 72];

    // Q fragments are loop-invariant: keep both c-halves in registers.
    const v16h aq0 = frag_a_f16(Qbase, HEADC, lane, 0);
    const v16h aq1 = frag_a_f16(Qbase, HEADC, lane, 32);

    v8f o0 = {}, o1 = {}, o2 = {}, o3 = {};
    float m[8], l[8];
#pragma unroll
    for (int i = 0; i < 8; ++i) { m[i] = -1e30f; l[i] = 0.0f; }

    for (int kb = 0; kb < SEQ_N; kb += 64) {
        __syncthreads();   // previous PV consumers done with Ks/Vt
        {   // stage K as [key][c], V transposed as [c][key]
            const int r = tid >> 2, c0 = (tid & 3) << 4;
            const v8h* ksrc = (const v8h*)(Kbase + (size_t)(kb + r) * HEADC + c0);
            *(v8h*)&Ks[r * 72 + c0]     = ksrc[0];
            *(v8h*)&Ks[r * 72 + c0 + 8] = ksrc[1];
            const _Float16* vsrc = Vbase + (size_t)(kb + r) * HEADC + c0;
#pragma unroll
            for (int i = 0; i < 16; ++i) Vt[(c0 + i) * 72 + r] = vsrc[i];
            // prefetch next key tile (speculative; OOB prefetch is dropped)
            __builtin_prefetch(Kbase + (size_t)(kb + 64 + r) * HEADC + c0, 0, 1);
        }
        __syncthreads();

        // S = Q K^T (log2 domain): 16 x 64 per wave, contraction over c = 64.
        v8f s[4] = {};
        {
            v16h bf[4];
#pragma unroll
            for (int nt = 0; nt < 4; ++nt)
                bf[nt] = frag_b_lds(&Ks[nt * 16 * 72], 72, lane);
#pragma unroll
            for (int nt = 0; nt < 4; ++nt) s[nt] = wmma_f16(aq0, bf[nt], s[nt]);
            __builtin_amdgcn_sched_group_barrier(SGB_DSREAD, 8, 0);
            __builtin_amdgcn_sched_group_barrier(SGB_WMMA,   4, 0);
#pragma unroll
            for (int nt = 0; nt < 4; ++nt)
                bf[nt] = frag_b_lds(&Ks[nt * 16 * 72 + 32], 72, lane);
#pragma unroll
            for (int nt = 0; nt < 4; ++nt) s[nt] = wmma_f16(aq1, bf[nt], s[nt]);
            __builtin_amdgcn_sched_group_barrier(SGB_DSREAD, 8, 0);
            __builtin_amdgcn_sched_group_barrier(SGB_WMMA,   4, 0);
        }

        // Row max across the 4 n-tiles, then across the 16-lane half.
        float mx[8];
#pragma unroll
        for (int i = 0; i < 8; ++i)
            mx[i] = fmaxf(fmaxf(s[0][i], s[1][i]), fmaxf(s[2][i], s[3][i]));
#pragma unroll
        for (int off = 8; off >= 1; off >>= 1)
#pragma unroll
            for (int i = 0; i < 8; ++i)
                mx[i] = fmaxf(mx[i], __shfl_xor(mx[i], off, 32));

        float alpha[8], rs[8];
#pragma unroll
        for (int i = 0; i < 8; ++i) {
            const float mn = fmaxf(m[i], mx[i]);
            alpha[i] = exp2f(m[i] - mn);
            m[i] = mn;
            rs[i] = 0.0f;
        }

        // P = exp2(S - m), stash to wave-private LDS row-major for A-frags.
#pragma unroll
        for (int nt = 0; nt < 4; ++nt)
#pragma unroll
            for (int i = 0; i < 8; ++i) {
                const float p = exp2f(s[nt][i] - m[i]);
                rs[i] += p;
                Pw[(i + ((lane & 16) ? 8 : 0)) * 72 + nt * 16 + (lane & 15)] =
                    (_Float16)p;
            }
#pragma unroll
        for (int off = 8; off >= 1; off >>= 1)
#pragma unroll
            for (int i = 0; i < 8; ++i) rs[i] += __shfl_xor(rs[i], off, 32);
#pragma unroll
        for (int i = 0; i < 8; ++i) {
            l[i] = l[i] * alpha[i] + rs[i];
            o0[i] *= alpha[i]; o1[i] *= alpha[i];
            o2[i] *= alpha[i]; o3[i] *= alpha[i];
        }
        __syncthreads();   // P visible; Vt stable

        // O += P @ V : contraction over the 64 keys of this block.
#pragma unroll
        for (int k0 = 0; k0 < 64; k0 += 32) {
            const v16h ap = frag_a_f16(Pw, 72, lane, k0);
            v16h bf[4];
#pragma unroll
            for (int ct = 0; ct < 4; ++ct)
                bf[ct] = frag_b_lds(&Vt[ct * 16 * 72 + k0], 72, lane);
            o0 = wmma_f16(ap, bf[0], o0);
            o1 = wmma_f16(ap, bf[1], o1);
            o2 = wmma_f16(ap, bf[2], o2);
            o3 = wmma_f16(ap, bf[3], o3);
            // 10 DS reads (2 for the P A-frag + 8 for V B-frags), then 4 WMMAs.
            __builtin_amdgcn_sched_group_barrier(SGB_DSREAD, 10, 0);
            __builtin_amdgcn_sched_group_barrier(SGB_WMMA,    4, 0);
        }
    }

    // Epilogue: normalize by l, store f16 into O[B, N, C] with C = h*64 + cc.
    const int col0 = h * HEADC + (lane & 15);
#pragma unroll
    for (int i = 0; i < 8; ++i) {
        const int qrow = qb + i + ((lane & 16) ? 8 : 0);
        const float inv = 1.0f / l[i];
        const size_t rowoff = ((size_t)b * SEQ_N + qrow) * DIMC;
        O[rowoff + col0 +  0] = (_Float16)(o0[i] * inv);
        O[rowoff + col0 + 16] = (_Float16)(o1[i] * inv);
        O[rowoff + col0 + 32] = (_Float16)(o2[i] * inv);
        O[rowoff + col0 + 48] = (_Float16)(o3[i] * inv);
    }
}

// ---------------------------------------------------------------------------
// Kernel 3: output projection. out = O @ wo.T + bo, f32 result.
// Same double-buffered tiling as kernel 1; A operand is already f16.
// ---------------------------------------------------------------------------
__global__ __launch_bounds__(256)
void out_gemm_kernel(const _Float16* __restrict__ Oin,
                     const float* __restrict__ wo, const float* __restrict__ bo,
                     float* __restrict__ out) {
    __shared__ _Float16 Bs[2][64 * 40];
    const int tid = threadIdx.x, lane = tid & 31, w = tid >> 5;
    const int m0 = blockIdx.x * 128, n0 = blockIdx.y * 64;
    const _Float16* Abase = Oin + (size_t)(m0 + w * 16) * DIMC;

    const int n = tid >> 2, koff = (tid & 3) << 3;
    const float* wsrc = wo + (size_t)(n0 + n) * DIMC + koff;
    _Float16* const bdst = &Bs[0][0] + n * 40 + koff;

    {
        const float4* s4 = (const float4*)wsrc;
        cvt_store_v8h(bdst, s4[0], s4[1]);
    }

    v8f acc[4] = {};
    int cur = 0;
    for (int k0 = 0; k0 < DIMC; k0 += 32) {
        __syncthreads();
        const bool more = (k0 + 32) < DIMC;
        float4 u0, u1;
        if (more) {
            const float4* s4 = (const float4*)(wsrc + k0 + 32);
            u0 = s4[0]; u1 = s4[1];
        }
        v16h a = frag_a_f16(Abase, DIMC, lane, k0);
        v16h bf[4];
#pragma unroll
        for (int nt = 0; nt < 4; ++nt)
            bf[nt] = frag_b_lds(&Bs[cur][nt * 16 * 40], 40, lane);
#pragma unroll
        for (int nt = 0; nt < 4; ++nt)
            acc[nt] = wmma_f16(a, bf[nt], acc[nt]);
        __builtin_amdgcn_sched_group_barrier(SGB_DSREAD, 8, 0);
        __builtin_amdgcn_sched_group_barrier(SGB_WMMA,   4, 0);
        if (more) cvt_store_v8h(bdst + (cur ^ 1) * 64 * 40, u0, u1);
        cur ^= 1;
    }

#pragma unroll
    for (int nt = 0; nt < 4; ++nt) {
        const int o = n0 + nt * 16 + (lane & 15);
        const float bval = bo[o];
#pragma unroll
        for (int i = 0; i < 8; ++i) {
            const int mrow = m0 + w * 16 + i + ((lane & 16) ? 8 : 0);
            out[(size_t)mrow * DIMC + o] = acc[nt][i] + bval;
        }
    }
}

// ---------------------------------------------------------------------------
extern "C" void kernel_launch(void* const* d_in, const int* in_sizes, int n_in,
                              void* d_out, int out_size, void* d_ws, size_t ws_size,
                              hipStream_t stream) {
    const float* x  = (const float*)d_in[0];
    const float* wq = (const float*)d_in[1];
    const float* bq = (const float*)d_in[2];
    const float* wk = (const float*)d_in[3];
    const float* bk = (const float*)d_in[4];
    const float* wv = (const float*)d_in[5];
    const float* bv = (const float*)d_in[6];
    const float* wo = (const float*)d_in[7];
    const float* bo = (const float*)d_in[8];

    const size_t PH = (size_t)NBATCH * NHEADS * SEQ_N * HEADC;  // 8388608
    _Float16* Q = (_Float16*)d_ws;
    _Float16* K = Q + PH;
    _Float16* V = K + PH;
    _Float16* O = V + PH;

    const int M = NBATCH * SEQ_N;                 // 8192
    dim3 blk(256);

    dim3 g1(M / 128, DIMC / 64, 3);               // 64 x 16 x 3
    qkv_gemm_kernel<<<g1, blk, 0, stream>>>(x, wq, bq, wk, bk, wv, bv, Q, K, V);

    dim3 g2(SEQ_N / 128, NBATCH * NHEADS);        // 16 x 64
    attn_kernel<<<g2, blk, 0, stream>>>(Q, K, V, O);

    dim3 g3(M / 128, DIMC / 64);                  // 64 x 16
    out_gemm_kernel<<<g3, blk, 0, stream>>>(O, wo, bo, (float*)d_out);
}
